// DecoderUnit_44513041056437
// MI455X (gfx1250) — matrix operand order repaired and checked by reading, not compiled, not run on hardware
//
#include <hip/hip_runtime.h>
#include <math.h>

// ---------------- problem constants (from reference) ----------------
#define Bc    4
#define Sc    1024
#define Ec    1024
#define NHc   16
#define HDc   64
#define FFc   2048
#define Rc    8
#define NTOK  (Bc * Sc)          // 4096 tokens
#define E3    (3 * Ec)           // 3072

// ---------------- CDNA5 async global->LDS path (guarded) ----------------
#if defined(__has_builtin)
#if __has_builtin(__builtin_amdgcn_global_load_async_to_lds_b128) && \
    __has_builtin(__builtin_amdgcn_s_wait_asynccnt)
#define HAVE_ASYNC_LDS 1
#endif
#endif

#if defined(HAVE_ASYNC_LDS)
typedef int v4i_ __attribute__((ext_vector_type(4)));
typedef __attribute__((address_space(1))) v4i_* gptr_b128;   // global int4*
typedef __attribute__((address_space(3))) v4i_* lptr_b128;   // LDS int4*
#endif

// ---------------- WMMA fragment types ----------------
typedef __attribute__((ext_vector_type(16))) __bf16 bf16x16;
typedef __attribute__((ext_vector_type(8)))  float  f32x8;

union FragU { bf16x16 v; unsigned u[8]; };

// fp32 -> bf16 (round to nearest even), pure integer path
__device__ inline unsigned short f2bf(float f) {
  unsigned u; __builtin_memcpy(&u, &f, 4);
  u += 0x7FFFu + ((u >> 16) & 1u);
  return (unsigned short)(u >> 16);
}
__device__ inline float bf2f(unsigned short h) {
  unsigned u = ((unsigned)h) << 16;
  float f; __builtin_memcpy(&f, &u, 4);
  return f;
}

// ---------------- generic bf16 WMMA GEMM with fused epilogue ----------------
// C[M,N] = op(A[M,K](bf16) @ W[K,N](f32->bf16) + bias) [*gate[row]] [+= Cf]
enum { F_GELU = 1, F_ACCUM = 2, F_STORE_BF = 4, F_STORE_F32 = 8, F_ROWSCALE = 16 };

#define BM 64
#define BN 64
#define BK 32
#define LDA_S 40   // 80-byte row stride: 16B-aligned per-lane B128 LDS destinations
#define LDB_S 34

__global__ __launch_bounds__(256)
void gemm_bf16_wmma(const unsigned short* __restrict__ A,
                    const float* __restrict__ W,
                    const float* __restrict__ bias,
                    float* __restrict__ Cf,
                    unsigned short* __restrict__ Cb,
                    const float* __restrict__ gate, int gateIdx,
                    int M, int N, int K, int flags) {
  __shared__ unsigned short ldsA[BM * LDA_S];   // [m][k] bf16
  __shared__ unsigned short ldsB[BN * LDB_S];   // [n][k] bf16 (transposed)

  const int tid  = threadIdx.x;
  const int lane = tid & 31, wave = tid >> 5;
  const int half = lane >> 4, l16 = lane & 15;
  const int m0 = blockIdx.y * BM;
  const int n0 = blockIdx.x * BN;

  const int mBase  = (wave & 3) * 16;
  const int nBase0 = (wave >> 2) * 32;
  const int nBase1 = nBase0 + 16;

  f32x8 c0 = {}; f32x8 c1 = {};

  const int ar = tid >> 2;            // 0..63 rows of A tile
  const int ac = (tid & 3) * 8;       // k offset (x8 bf16 = 16B)
  const int wr = tid >> 3;            // 0..31 rows of W tile
  const int wc = (tid & 7) * 8;       // n offset

  for (int k0 = 0; k0 < K; k0 += BK) {
    // stage A tile (already bf16): async global->LDS B128 when available
#if defined(HAVE_ASYNC_LDS)
    __builtin_amdgcn_global_load_async_to_lds_b128(
        (gptr_b128)(const void*)(A + (size_t)(m0 + ar) * K + k0 + ac),
        (lptr_b128)(void*)(&ldsA[ar * LDA_S + ac]), 0, 0);
#else
    {
      const uint4 av = *reinterpret_cast<const uint4*>(A + (size_t)(m0 + ar) * K + k0 + ac);
      unsigned* dst = reinterpret_cast<unsigned*>(&ldsA[ar * LDA_S + ac]);
      dst[0] = av.x; dst[1] = av.y; dst[2] = av.z; dst[3] = av.w;
    }
#endif
    // stage W tile: fp32 -> bf16, transposed to [n][k]
    {
      const float* wp = W + (size_t)(k0 + wr) * N + n0 + wc;
      const float4 w0v = *reinterpret_cast<const float4*>(wp);
      const float4 w1v = *reinterpret_cast<const float4*>(wp + 4);
      float wv[8] = {w0v.x, w0v.y, w0v.z, w0v.w, w1v.x, w1v.y, w1v.z, w1v.w};
#pragma unroll
      for (int j = 0; j < 8; ++j)
        ldsB[(wc + j) * LDB_S + wr] = f2bf(wv[j]);
    }
    // speculative prefetch of next K-step's W tile (global_prefetch_b8)
    if (k0 + BK < K)
      __builtin_prefetch(W + (size_t)(k0 + BK + wr) * N + n0 + wc, 0, 1);
#if defined(HAVE_ASYNC_LDS)
    __builtin_amdgcn_s_wait_asynccnt(0);
#endif
    __syncthreads();

    // A fragment: 16x32 bf16 per ISA layout (lane<16: K 0-7/16-23, lane>=16: K 8-15/24-31)
    FragU a;
#pragma unroll
    for (int i = 0; i < 4; ++i) {
      a.u[i]     = *reinterpret_cast<unsigned*>(&ldsA[(mBase + l16) * LDA_S + half * 8 + 2 * i]);
      a.u[4 + i] = *reinterpret_cast<unsigned*>(&ldsA[(mBase + l16) * LDA_S + 16 + half * 8 + 2 * i]);
    }
    // B fragments: 32x16, per-lane column n, K pairs 8*half + 16*(v>>2) + 2*(v&3)
    FragU b0, b1;
#pragma unroll
    for (int v = 0; v < 8; ++v) {
      b0.u[v] = *reinterpret_cast<unsigned*>(&ldsB[(nBase0 + l16) * LDB_S + half * 8 + (v >> 2) * 16 + 2 * (v & 3)]);
      b1.u[v] = *reinterpret_cast<unsigned*>(&ldsB[(nBase1 + l16) * LDB_S + half * 8 + (v >> 2) * 16 + 2 * (v & 3)]);
    }
    c0 = __builtin_amdgcn_wmma_f32_16x16x32_bf16(false, a.v, false, b0.v, (short)0, c0, false, false);
    c1 = __builtin_amdgcn_wmma_f32_16x16x32_bf16(false, a.v, false, b1.v, (short)0, c1, false, false);
    __syncthreads();
  }

  // fused epilogue; C/D layout: VGPR r -> row r + 8*half, col = lane&15
#pragma unroll
  for (int t = 0; t < 2; ++t) {
    f32x8 c = t ? c1 : c0;
    const int nb = t ? nBase1 : nBase0;
#pragma unroll
    for (int r = 0; r < 8; ++r) {
      const int row = m0 + mBase + r + 8 * half;
      const int col = n0 + nb + l16;
      float v = c[r] + bias[col];
      if (flags & F_GELU)     v = 0.5f * v * (1.0f + erff(v * 0.70710678118654752f));
      if (flags & F_ROWSCALE) v *= gate[row * Rc + gateIdx];
      const size_t off = (size_t)row * N + col;
      if (flags & F_STORE_F32) {
        if (flags & F_ACCUM) Cf[off] += v; else Cf[off] = v;
      }
      if (flags & F_STORE_BF) Cb[off] = f2bf(v);
    }
  }
}

// ---------------- flash attention (causal) with WMMA ----------------
// one block per (qblk of 64 queries, head, batch); 8 waves
__global__ __launch_bounds__(256)
void attn_wmma(const float* __restrict__ qkv, unsigned short* __restrict__ obf) {
  __shared__ unsigned short ldsQ[64 * 66];   // [m][d] bf16
  __shared__ unsigned short ldsK[32 * 66];   // [key][d] bf16 (B-layout for QK^T)
  __shared__ unsigned short ldsV[64 * 34];   // [d][key] bf16 (B-layout for PV)
  __shared__ float          s_lds[64 * 33];  // fp32 scores
  __shared__ unsigned short p_lds[64 * 34];  // probs bf16
  __shared__ float m_s[64], l_s[64], sc_s[64];

  const int tid  = threadIdx.x;
  const int lane = tid & 31, wave = tid >> 5;
  const int half = lane >> 4, l16 = lane & 15;
  const int qblk = blockIdx.x, h = blockIdx.y, b = blockIdx.z;

  // load 64x64 Q block
  {
    const int m = tid >> 2, d0 = (tid & 3) * 16;
    const float* src = qkv + (size_t)(b * Sc + qblk * 64 + m) * E3 + h * HDc + d0;
#pragma unroll
    for (int j = 0; j < 16; ++j) ldsQ[m * 66 + d0 + j] = f2bf(src[j]);
  }
  if (tid < 64) { m_s[tid] = -1e30f; l_s[tid] = 0.0f; }

  f32x8 o0 = {}; f32x8 o1 = {};
  const int mT_s = wave >> 1, nT_s = wave & 1;   // S = QK^T tiling (4x2 tiles)
  const int mT_o = wave & 3,  dP   = wave >> 2;  // O = PV tiling (4x(2x2) tiles)
  const int kbMax = qblk * 2 + 1;

  __syncthreads();
  for (int kb = 0; kb <= kbMax; ++kb) {
    // stage K (row-major [key][d]) and V transposed ([d][key])
    {
      const int key = tid >> 3, d0 = (tid & 7) * 8;
      const float* kp = qkv + (size_t)(b * Sc + kb * 32 + key) * E3 + Ec + h * HDc + d0;
      const float* vp = kp + Ec;
#pragma unroll
      for (int j = 0; j < 8; ++j) {
        ldsK[key * 66 + d0 + j]   = f2bf(kp[j]);
        ldsV[(d0 + j) * 34 + key] = f2bf(vp[j]);
      }
    }
    __syncthreads();

    // S tile = Q @ K^T  (contract over d = 2 chunks of 32)
    {
      f32x8 s = {};
#pragma unroll
      for (int dk = 0; dk < 64; dk += 32) {
        FragU a, bk;
#pragma unroll
        for (int i = 0; i < 4; ++i) {
          a.u[i]     = *reinterpret_cast<unsigned*>(&ldsQ[(mT_s * 16 + l16) * 66 + dk + half * 8 + 2 * i]);
          a.u[4 + i] = *reinterpret_cast<unsigned*>(&ldsQ[(mT_s * 16 + l16) * 66 + dk + 16 + half * 8 + 2 * i]);
        }
#pragma unroll
        for (int v = 0; v < 8; ++v)
          bk.u[v] = *reinterpret_cast<unsigned*>(&ldsK[(nT_s * 16 + l16) * 66 + dk + half * 8 + (v >> 2) * 16 + 2 * (v & 3)]);
        s = __builtin_amdgcn_wmma_f32_16x16x32_bf16(false, a.v, false, bk.v, (short)0, s, false, false);
      }
      // scale 1/sqrt(64) + causal mask, spill to LDS
#pragma unroll
      for (int r = 0; r < 8; ++r) {
        const int qm = mT_s * 16 + r + 8 * half;
        const int kn = nT_s * 16 + l16;
        const int qg = qblk * 64 + qm, kg = kb * 32 + kn;
        s_lds[qm * 33 + kn] = (kg <= qg) ? (s[r] * 0.125f) : -1e30f;
      }
    }
    __syncthreads();

    // online softmax, one row per thread
    if (tid < 64) {
      const float mOld = m_s[tid];
      float mNew = mOld;
#pragma unroll 4
      for (int j = 0; j < 32; ++j) mNew = fmaxf(mNew, s_lds[tid * 33 + j]);
      const float scl = __expf(mOld - mNew);
      float lsum = 0.0f;
      for (int j = 0; j < 32; ++j) {
        const float p = __expf(s_lds[tid * 33 + j] - mNew);
        lsum += p;
        p_lds[tid * 34 + j] = f2bf(p);
      }
      m_s[tid]  = mNew;
      l_s[tid]  = l_s[tid] * scl + lsum;
      sc_s[tid] = scl;
    }
    __syncthreads();

    // O accumulators: rescale then O += P @ V
    {
      FragU a, b0, b1;
#pragma unroll
      for (int i = 0; i < 4; ++i) {
        a.u[i]     = *reinterpret_cast<unsigned*>(&p_lds[(mT_o * 16 + l16) * 34 + half * 8 + 2 * i]);
        a.u[4 + i] = *reinterpret_cast<unsigned*>(&p_lds[(mT_o * 16 + l16) * 34 + 16 + half * 8 + 2 * i]);
      }
      const int d0 = dP * 32, d1 = d0 + 16;
#pragma unroll
      for (int v = 0; v < 8; ++v) {
        b0.u[v] = *reinterpret_cast<unsigned*>(&ldsV[(d0 + l16) * 34 + half * 8 + (v >> 2) * 16 + 2 * (v & 3)]);
        b1.u[v] = *reinterpret_cast<unsigned*>(&ldsV[(d1 + l16) * 34 + half * 8 + (v >> 2) * 16 + 2 * (v & 3)]);
      }
#pragma unroll
      for (int r = 0; r < 8; ++r) {
        const float scl = sc_s[mT_o * 16 + r + 8 * half];
        o0[r] *= scl; o1[r] *= scl;
      }
      o0 = __builtin_amdgcn_wmma_f32_16x16x32_bf16(false, a.v, false, b0.v, (short)0, o0, false, false);
      o1 = __builtin_amdgcn_wmma_f32_16x16x32_bf16(false, a.v, false, b1.v, (short)0, o1, false, false);
    }
    __syncthreads();
  }

  // normalize and write bf16 O (layout [b,s,h*64+d])
#pragma unroll
  for (int r = 0; r < 8; ++r) {
    const int qm = mT_o * 16 + r + 8 * half;
    const float inv = 1.0f / l_s[qm];
    const int qg = qblk * 64 + qm;
    const int d0 = dP * 32 + l16, d1 = d0 + 16;
    const size_t base = (size_t)(b * Sc + qg) * Ec + h * HDc;
    obf[base + d0] = f2bf(o0[r] * inv);
    obf[base + d1] = f2bf(o1[r] * inv);
  }
}

// ---------------- fused residual + layernorm (optional bf16 copy) ----------------
__global__ __launch_bounds__(256)
void ln_kernel(const float* __restrict__ inp, const float* __restrict__ resid,
               const float* __restrict__ w, const float* __restrict__ bias,
               float* __restrict__ outf, unsigned short* __restrict__ outb) {
  __shared__ float red[256];
  __shared__ float red2[256];
  const int row = blockIdx.x, tid = threadIdx.x;
  float vals[4];
  float s = 0.0f;
#pragma unroll
  for (int i = 0; i < 4; ++i) {
    const int e = tid + i * 256;
    float v = inp[(size_t)row * Ec + e];
    if (resid) v += resid[(size_t)row * Ec + e];
    vals[i] = v; s += v;
  }
  red[tid] = s;
  __syncthreads();
  for (int off = 128; off > 0; off >>= 1) { if (tid < off) red[tid] += red[tid + off]; __syncthreads(); }
  const float mean = red[0] * (1.0f / Ec);
  float s2 = 0.0f;
#pragma unroll
  for (int i = 0; i < 4; ++i) { const float d = vals[i] - mean; s2 += d * d; }
  red2[tid] = s2;
  __syncthreads();
  for (int off = 128; off > 0; off >>= 1) { if (tid < off) red2[tid] += red2[tid + off]; __syncthreads(); }
  const float inv = rsqrtf(red2[0] * (1.0f / Ec) + 1e-5f);
#pragma unroll
  for (int i = 0; i < 4; ++i) {
    const int e = tid + i * 256;
    const float o = (vals[i] - mean) * inv * w[e] + bias[e];
    if (outf) outf[(size_t)row * Ec + e] = o;
    if (outb) outb[(size_t)row * Ec + e] = f2bf(o);
  }
}

// ---------------- routing: softmax sim -> top2(+bias) -> renorm gate ----------------
__global__ void route_kernel(const unsigned short* __restrict__ xb,
                             const float* __restrict__ centroid,
                             const float* __restrict__ rbias,
                             float* __restrict__ gate) {
  const int t = blockIdx.x * blockDim.x + threadIdx.x;
  if (t >= NTOK) return;
  float dots[Rc] = {};
  for (int e = 0; e < Ec; ++e) {
    const float x = bf2f(xb[(size_t)t * Ec + e]);
#pragma unroll
    for (int r = 0; r < Rc; ++r) dots[r] += x * centroid[r * Ec + e];
  }
  float mx = dots[0];
#pragma unroll
  for (int r = 1; r < Rc; ++r) mx = fmaxf(mx, dots[r]);
  float sim[Rc]; float sum = 0.0f;
#pragma unroll
  for (int r = 0; r < Rc; ++r) { sim[r] = __expf(dots[r] - mx); sum += sim[r]; }
  const float isum = 1.0f / sum;
#pragma unroll
  for (int r = 0; r < Rc; ++r) sim[r] *= isum;
  int i1 = 0; float v1 = -1e30f;
#pragma unroll
  for (int r = 0; r < Rc; ++r) { const float v = sim[r] + rbias[r]; if (v > v1) { v1 = v; i1 = r; } }
  int i2 = 0; float v2 = -1e30f;
#pragma unroll
  for (int r = 0; r < Rc; ++r) { if (r == i1) continue; const float v = sim[r] + rbias[r]; if (v > v2) { v2 = v; i2 = r; } }
  const float denom = 1.0f / (sim[i1] + sim[i2]);
#pragma unroll
  for (int r = 0; r < Rc; ++r) gate[(size_t)t * Rc + r] = 0.0f;
  gate[(size_t)t * Rc + i1] = sim[i1] * denom;
  gate[(size_t)t * Rc + i2] = sim[i2] * denom;
}

// ---------------- fp32 -> bf16 conversion ----------------
__global__ void cvt_kernel(const float* __restrict__ in, unsigned short* __restrict__ out, int n) {
  const int i = blockIdx.x * blockDim.x + threadIdx.x;
  if (i < n) out[i] = f2bf(in[i]);
}

// ---------------- orchestration ----------------
extern "C" void kernel_launch(void* const* d_in, const int* in_sizes, int n_in,
                              void* d_out, int out_size, void* d_ws, size_t ws_size,
                              hipStream_t stream) {
  const float* x        = (const float*)d_in[0];
  const float* qkv_w    = (const float*)d_in[1];
  const float* qkv_b    = (const float*)d_in[2];
  const float* out_w    = (const float*)d_in[3];
  const float* out_b    = (const float*)d_in[4];
  const float* norm1_w  = (const float*)d_in[5];
  const float* norm1_b  = (const float*)d_in[6];
  const float* sw1      = (const float*)d_in[7];
  const float* sb1      = (const float*)d_in[8];
  const float* sw2      = (const float*)d_in[9];
  const float* sb2      = (const float*)d_in[10];
  const float* rw1      = (const float*)d_in[11];
  const float* rb1      = (const float*)d_in[12];
  const float* rw2      = (const float*)d_in[13];
  const float* rb2      = (const float*)d_in[14];
  const float* centroid = (const float*)d_in[15];
  const float* rbias    = (const float*)d_in[16];
  const float* norm2_w  = (const float*)d_in[17];
  const float* norm2_b  = (const float*)d_in[18];

  // workspace layout (bytes)
  char* ws = (char*)d_ws;
  const size_t SZ_QKV  = (size_t)NTOK * E3 * 4;   // 48 MB (freed after attention)
  const size_t SZ_OBF  = (size_t)NTOK * Ec * 2;   //  8 MB
  const size_t SZ_YATT = (size_t)NTOK * Ec * 4;   // 16 MB
  const size_t SZ_XB   = (size_t)NTOK * Ec * 2;   //  8 MB

  float*          qkv_buf = (float*)ws;
  unsigned short* o_bf    = (unsigned short*)(ws + SZ_QKV);
  float*          y_attn  = (float*)(ws + SZ_QKV + SZ_OBF);
  unsigned short* xb      = (unsigned short*)(ws + SZ_QKV + SZ_OBF + SZ_YATT);
  unsigned short* x1b     = (unsigned short*)(ws + SZ_QKV + SZ_OBF + SZ_YATT + SZ_XB);
  // MoE scratch aliases the qkv buffer (qkv dead after attention)
  float*          y_buf   = (float*)ws;                                       // 16 MB
  unsigned short* hb      = (unsigned short*)(ws + (size_t)NTOK * Ec * 4);    // 16 MB
  float*          gate    = (float*)(ws + (size_t)NTOK * Ec * 4 + (size_t)NTOK * FFc * 2);

  const dim3 blk(256);

  // 1) x -> bf16
  cvt_kernel<<<(NTOK * Ec) / 256, blk, 0, stream>>>(x, xb, NTOK * Ec);

  // 2) qkv = x @ qkv_w + b   [4096 x 3072 x 1024]
  gemm_bf16_wmma<<<dim3(E3 / BN, NTOK / BM), blk, 0, stream>>>(
      xb, qkv_w, qkv_b, qkv_buf, nullptr, nullptr, 0, NTOK, E3, Ec, F_STORE_F32);

  // 3) causal flash attention -> o (bf16)
  attn_wmma<<<dim3(Sc / 64, NHc, Bc), blk, 0, stream>>>(qkv_buf, o_bf);

  // 4) y_attn = o @ out_w + out_b   [4096 x 1024 x 1024]
  gemm_bf16_wmma<<<dim3(Ec / BN, NTOK / BM), blk, 0, stream>>>(
      o_bf, out_w, out_b, y_attn, nullptr, nullptr, 0, NTOK, Ec, Ec, F_STORE_F32);

  // 5) x1 = LN(x + y_attn) -> y_buf (f32 accumulator) + x1b (bf16)
  ln_kernel<<<NTOK, blk, 0, stream>>>(y_attn, x, norm1_w, norm1_b, y_buf, x1b);

  // 6) routing gates
  route_kernel<<<NTOK / 256, blk, 0, stream>>>(x1b, centroid, rbias, gate);

  // 7) shared expert: h = gelu(x1 @ sw1 + sb1) ; y += h @ sw2 + sb2
  gemm_bf16_wmma<<<dim3(FFc / BN, NTOK / BM), blk, 0, stream>>>(
      x1b, sw1, sb1, nullptr, hb, nullptr, 0, NTOK, FFc, Ec, F_GELU | F_STORE_BF);
  gemm_bf16_wmma<<<dim3(Ec / BN, NTOK / BM), blk, 0, stream>>>(
      hb, sw2, sb2, y_buf, nullptr, nullptr, 0, NTOK, Ec, FFc, F_ACCUM | F_STORE_F32);

  // 8) routed experts (dense loop, gated accumulate)
  for (int i = 0; i < Rc; ++i) {
    gemm_bf16_wmma<<<dim3(FFc / BN, NTOK / BM), blk, 0, stream>>>(
        x1b, rw1 + (size_t)i * Ec * FFc, rb1 + (size_t)i * FFc,
        nullptr, hb, nullptr, 0, NTOK, FFc, Ec, F_GELU | F_STORE_BF);
    gemm_bf16_wmma<<<dim3(Ec / BN, NTOK / BM), blk, 0, stream>>>(
        hb, rw2 + (size_t)i * FFc * Ec, rb2 + (size_t)i * Ec,
        y_buf, nullptr, gate, i, NTOK, Ec, FFc, F_ACCUM | F_ROWSCALE | F_STORE_F32);
  }

  // 9) out = LN(y)
  ln_kernel<<<NTOK, blk, 0, stream>>>(y_buf, nullptr, norm2_w, norm2_b, (float*)d_out, nullptr);
}
